// MambaIncidentModel_79912161509559
// MI455X (gfx1250) — matrix operand order repaired and checked by reading, not compiled, not run on hardware
//
#include <hip/hip_runtime.h>
#include <hip/hip_bf16.h>
#include <math.h>

// ---------------- model dims ----------------
#define BB   8
#define TT   1024
#define FF   64
#define DM   256
#define LL   4
#define DI   512
#define NN   16
#define DCV  4
#define DTR  16
#define XPE  48   // DTR + 2N

typedef __attribute__((ext_vector_type(16))) _Float16 v16h;
typedef __attribute__((ext_vector_type(8)))  _Float16 v8h;
typedef __attribute__((ext_vector_type(8)))  float    v8f;

// ---------------- helpers ----------------
__device__ __forceinline__ float dev_silu(float x) {
    return x / (1.0f + __expf(-x));
}
__device__ __forceinline__ float dev_softplus(float x) {
    return (x > 20.0f) ? x : __logf(1.0f + __expf(x));
}

// A-fragment (16x32, MxK): lane m = l&15, half h = l>>4 holds
// K = {8h..8h+7} (vgpr0..3) and {16+8h..16+8h+7} (vgpr4..7).
template <bool GUARD>
__device__ __forceinline__ v16h load_fragA(const _Float16* __restrict__ rowp,
                                           int kk, int hh, int kvalid) {
    int c0 = kk + 8 * hh;
    int c1 = kk + 16 + 8 * hh;
    v8h lo, hi;
    if (GUARD) {
        v8h z = {};
        lo = (c0 < kvalid) ? *(const v8h*)(rowp + c0) : z;
        hi = (c1 < kvalid) ? *(const v8h*)(rowp + c1) : z;
    } else {
        lo = *(const v8h*)(rowp + c0);
        hi = *(const v8h*)(rowp + c1);
    }
    return __builtin_shufflevector(lo, hi, 0,1,2,3,4,5,6,7,8,9,10,11,12,13,14,15);
}

// B-fragment (32x16, KxN): lane n = l&15, half h = l>>4 holds K = {8h..8h+7}
// and {16+8h..16+8h+7}. Weights are (out,in) row-major, so row n gives the column.
template <bool GUARD>
__device__ __forceinline__ v16h load_fragB(const _Float16* __restrict__ rowp,
                                           int kk, int hh, int kvalid) {
    int c0 = kk + 8 * hh;
    int c1 = kk + 16 + 8 * hh;
    v8h lo, hi;
    if (GUARD) {
        v8h z = {};
        lo = (c0 < kvalid) ? *(const v8h*)(rowp + c0) : z;
        hi = (c1 < kvalid) ? *(const v8h*)(rowp + c1) : z;
    } else {
        lo = *(const v8h*)(rowp + c0);
        hi = *(const v8h*)(rowp + c1);
    }
    return __builtin_shufflevector(lo, hi, 0,1,2,3,4,5,6,7,8,9,10,11,12,13,14,15);
}

// ---------------- generic WMMA GEMM ----------------
// C[M x Ntiles*16] = A[M x K] * Bw[Ntiles*16 x K]^T   (Bw row-major out x in)
// Each wave owns a 32(M) x 32(N) register tile: 2 A-frags + 2 B-frags per
// K-step feed 4 WMMAs (16 f16-FLOP per loaded byte). The second n-tile has a
// wave-uniform validity guard (odd Ntiles), so EXEC is all-ones around WMMA.
// EPI: 0 = none, 1 = +bias, 2 = softplus(x + bias).
template <bool GUARD, int EPI, bool HAS16>
__global__ __launch_bounds__(128)
void gemm_f16_wmma(const _Float16* __restrict__ A, int lda,
                   const _Float16* __restrict__ Bw, int ldb,
                   float* __restrict__ C, _Float16* __restrict__ C16, int ldc,
                   int Ntiles, int Kpad, int Kvalid,
                   const float* __restrict__ bias) {
    const int wave = threadIdx.x >> 5;
    const int lane = threadIdx.x & 31;
    const int mt   = blockIdx.x;                     // 32-row stripe index
    const int nt0  = (blockIdx.y * 4 + wave) * 2;    // first of 2 n-tiles
    if (nt0 >= Ntiles) return;                       // wave-uniform
    const bool nt1v = (nt0 + 1) < Ntiles;            // wave-uniform

    const int m  = lane & 15;                        // row (A) / col (B,D)
    const int hh = lane >> 4;

    const _Float16* arow0 = A  + (size_t)(mt * 32 + m) * lda;
    const _Float16* arow1 = arow0 + (size_t)16 * lda;
    const _Float16* brow0 = Bw + (size_t)(nt0 * 16 + m) * ldb;
    const _Float16* brow1 = brow0 + (size_t)16 * ldb;

    v8f acc00 = {}, acc10 = {}, acc01 = {}, acc11 = {};

    if (nt1v) {
        for (int kk = 0; kk < Kpad; kk += 32) {
            if (!GUARD && (kk + 32 < Kpad)) {
                __builtin_prefetch(arow0 + kk + 32, 0, 3);
                __builtin_prefetch(arow1 + kk + 32, 0, 3);
                __builtin_prefetch(brow0 + kk + 32, 0, 3);
                __builtin_prefetch(brow1 + kk + 32, 0, 3);
            }
            v16h bf0 = load_fragB<GUARD>(brow0, kk, hh, Kvalid);
            v16h bf1 = load_fragB<GUARD>(brow1, kk, hh, Kvalid);
            v16h af0 = load_fragA<GUARD>(arow0, kk, hh, Kvalid);
            v16h af1 = load_fragA<GUARD>(arow1, kk, hh, Kvalid);
            acc00 = __builtin_amdgcn_wmma_f32_16x16x32_f16(
                        false, af0, false, bf0, (short)0, acc00, false, false);
            acc10 = __builtin_amdgcn_wmma_f32_16x16x32_f16(
                        false, af1, false, bf0, (short)0, acc10, false, false);
            acc01 = __builtin_amdgcn_wmma_f32_16x16x32_f16(
                        false, af0, false, bf1, (short)0, acc01, false, false);
            acc11 = __builtin_amdgcn_wmma_f32_16x16x32_f16(
                        false, af1, false, bf1, (short)0, acc11, false, false);
        }
    } else {
        for (int kk = 0; kk < Kpad; kk += 32) {
            if (!GUARD && (kk + 32 < Kpad)) {
                __builtin_prefetch(arow0 + kk + 32, 0, 3);
                __builtin_prefetch(arow1 + kk + 32, 0, 3);
                __builtin_prefetch(brow0 + kk + 32, 0, 3);
            }
            v16h bf0 = load_fragB<GUARD>(brow0, kk, hh, Kvalid);
            v16h af0 = load_fragA<GUARD>(arow0, kk, hh, Kvalid);
            v16h af1 = load_fragA<GUARD>(arow1, kk, hh, Kvalid);
            acc00 = __builtin_amdgcn_wmma_f32_16x16x32_f16(
                        false, af0, false, bf0, (short)0, acc00, false, false);
            acc10 = __builtin_amdgcn_wmma_f32_16x16x32_f16(
                        false, af1, false, bf0, (short)0, acc10, false, false);
        }
    }

    // ---- epilogue ----
    const int col0 = nt0 * 16 + m;
    const int col1 = col0 + 16;
    const float bv0 = (EPI >= 1) ? bias[col0] : 0.0f;
    const float bv1 = (EPI >= 1 && nt1v) ? bias[col1] : 0.0f;
#pragma unroll
    for (int r = 0; r < 8; ++r) {
        int row0 = mt * 32 + r + 8 * hh;
        int row1 = row0 + 16;
        float v00 = acc00[r], v10 = acc10[r];
        if (EPI >= 1) { v00 += bv0; v10 += bv0; }
        if (EPI == 2) { v00 = dev_softplus(v00); v10 = dev_softplus(v10); }
        C[(size_t)row0 * ldc + col0] = v00;
        C[(size_t)row1 * ldc + col0] = v10;
        if (HAS16) {
            C16[(size_t)row0 * ldc + col0] = (_Float16)v00;
            C16[(size_t)row1 * ldc + col0] = (_Float16)v10;
        }
        if (nt1v) {
            float v01 = acc01[r], v11 = acc11[r];
            if (EPI >= 1) { v01 += bv1; v11 += bv1; }
            if (EPI == 2) { v01 = dev_softplus(v01); v11 = dev_softplus(v11); }
            C[(size_t)row0 * ldc + col1] = v01;
            C[(size_t)row1 * ldc + col1] = v11;
            if (HAS16) {
                C16[(size_t)row0 * ldc + col1] = (_Float16)v01;
                C16[(size_t)row1 * ldc + col1] = (_Float16)v11;
            }
        }
    }
}

// ---------------- fp32 -> f16 convert ----------------
__global__ void cvt_f32_to_f16(const float* __restrict__ in,
                               _Float16* __restrict__ out, int n) {
    int i = blockIdx.x * blockDim.x + threadIdx.x;
    if (i < n) out[i] = (_Float16)in[i];
}

// ---------------- depthwise causal conv (DC=4) + silu ----------------
// xz layout: [B*T, 2*DI]; conv over first DI channels.
__global__ void conv_silu_kernel(const float* __restrict__ xz,
                                 const float* __restrict__ cw,  // [DI, DC]
                                 const float* __restrict__ cb,  // [DI]
                                 float* __restrict__ xc,
                                 _Float16* __restrict__ xc16) {
    int idx = blockIdx.x * blockDim.x + threadIdx.x;
    if (idx >= BB * TT * DI) return;
    int d = idx & (DI - 1);
    int t = (idx >> 9) & (TT - 1);
    int b = idx >> 19;
    float s = cb[d];
#pragma unroll
    for (int k = 0; k < DCV; ++k) {
        int tt = t - (DCV - 1) + k;
        if (tt >= 0)
            s += xz[((size_t)(b * TT + tt)) * (2 * DI) + d] * cw[d * DCV + k];
    }
    s = dev_silu(s);
    xc[idx]   = s;
    xc16[idx] = (_Float16)s;
}

// ---------------- selective scan ----------------
// One block per batch element; thread d owns the (d, n=0..15) state row.
// B_t / C_t (32 floats, shared by all d) broadcast through LDS.
__global__ __launch_bounds__(512)
void scan_kernel(const float* __restrict__ dt,    // [B*T, DI]
                 const float* __restrict__ xdbl,  // [B*T, 48]
                 const float* __restrict__ xc,    // [B*T, DI]
                 const float* __restrict__ xz,    // [B*T, 2*DI] (z in cols DI..)
                 const float* __restrict__ Alog,  // [DI, N] (layer slice)
                 const float* __restrict__ Dp,    // [DI]
                 _Float16* __restrict__ y16) {    // [B*T, DI]
    const int b = blockIdx.x;
    const int d = threadIdx.x;

    float Av[NN];
#pragma unroll
    for (int n = 0; n < NN; ++n) Av[n] = -__expf(Alog[d * NN + n]);
    const float Dv = Dp[d];

    float hst[NN];
#pragma unroll
    for (int n = 0; n < NN; ++n) hst[n] = 0.0f;

    __shared__ float sBC[2 * NN];
    for (int t = 0; t < TT; ++t) {
        size_t tok = (size_t)(b * TT + t);
        if (threadIdx.x < 2 * NN)
            sBC[threadIdx.x] = xdbl[tok * XPE + DTR + threadIdx.x];
        __syncthreads();

        float dtv = dt[tok * DI + d];
        float u   = xc[tok * DI + d];
        float du  = dtv * u;
        float y   = 0.0f;
#pragma unroll
        for (int n = 0; n < NN; ++n) {
            hst[n] = __expf(dtv * Av[n]) * hst[n] + du * sBC[n];
            y += hst[n] * sBC[NN + n];
        }
        float zv = xz[tok * (2 * DI) + DI + d];
        y16[tok * DI + d] = (_Float16)((y + Dv * u) * dev_silu(zv));
        __syncthreads();
    }
}

// ---------------- layernorm(last token) + head ----------------
__global__ __launch_bounds__(DM)
void ln_head_kernel(const float* __restrict__ h,      // [B*T, DM]
                    const float* __restrict__ ln_g,
                    const float* __restrict__ ln_b,
                    const float* __restrict__ head_w, // [DM]
                    const float* __restrict__ head_b, // [1]
                    float* __restrict__ out) {        // [B]
    const int b = blockIdx.x;
    const int i = threadIdx.x;
    __shared__ float red[DM];

    float v = h[((size_t)b * TT + (TT - 1)) * DM + i];

    red[i] = v;
    __syncthreads();
    for (int s = DM / 2; s > 0; s >>= 1) {
        if (i < s) red[i] += red[i + s];
        __syncthreads();
    }
    float mu = red[0] / DM;
    __syncthreads();

    float c = v - mu;
    red[i] = c * c;
    __syncthreads();
    for (int s = DM / 2; s > 0; s >>= 1) {
        if (i < s) red[i] += red[i + s];
        __syncthreads();
    }
    float var = red[0] / DM;
    __syncthreads();

    float hn = c * rsqrtf(var + 1e-5f) * ln_g[i] + ln_b[i];
    red[i] = hn * head_w[i];
    __syncthreads();
    for (int s = DM / 2; s > 0; s >>= 1) {
        if (i < s) red[i] += red[i + s];
        __syncthreads();
    }
    if (i == 0) out[b] = red[0] + head_b[0];
}

// ---------------- host launcher ----------------
extern "C" void kernel_launch(void* const* d_in, const int* in_sizes, int n_in,
                              void* d_out, int out_size, void* d_ws, size_t ws_size,
                              hipStream_t stream) {
    (void)in_sizes; (void)n_in; (void)out_size; (void)ws_size;

    const float* x       = (const float*)d_in[0];   // [B,T,F]
    const float* proj_w  = (const float*)d_in[1];   // [DM,F]
    const float* proj_b  = (const float*)d_in[2];   // [DM]
    const float* ipw     = (const float*)d_in[3];   // [L, 2DI, DM]
    const float* conv_w  = (const float*)d_in[4];   // [L, DI, DC]
    const float* conv_b  = (const float*)d_in[5];   // [L, DI]
    const float* xpw     = (const float*)d_in[6];   // [L, 48, DI]
    const float* dtw     = (const float*)d_in[7];   // [L, DI, DTR]
    const float* dtb     = (const float*)d_in[8];   // [L, DI]
    const float* Alog    = (const float*)d_in[9];   // [L, DI, N]
    const float* Dp      = (const float*)d_in[10];  // [L, DI]
    const float* opw     = (const float*)d_in[11];  // [L, DM, DI]
    const float* ln_g    = (const float*)d_in[12];
    const float* ln_b    = (const float*)d_in[13];
    const float* head_w  = (const float*)d_in[14];
    const float* head_b  = (const float*)d_in[15];
    float* out = (float*)d_out;

    const size_t BT = (size_t)BB * TT;

    // workspace carve-up (256B aligned)
    char* base = (char*)d_ws;
    size_t off = 0;
    auto carve = [&](size_t bytes) -> char* {
        char* p = base + off;
        off = (off + bytes + 255) & ~(size_t)255;
        return p;
    };
    _Float16* x16     = (_Float16*)carve(BT * FF * 2);
    _Float16* wproj16 = (_Float16*)carve((size_t)DM * FF * 2);
    _Float16* win16   = (_Float16*)carve((size_t)LL * 2 * DI * DM * 2);
    _Float16* wxp16   = (_Float16*)carve((size_t)LL * XPE * DI * 2);
    _Float16* wdt16   = (_Float16*)carve((size_t)LL * DI * DTR * 2);
    _Float16* wout16  = (_Float16*)carve((size_t)LL * DM * DI * 2);
    float*    hbuf    = (float*)   carve(BT * DM * 4);
    _Float16* h16     = (_Float16*)carve(BT * DM * 2);
    float*    xzbuf   = (float*)   carve(BT * 2 * DI * 4);
    float*    xcbuf   = (float*)   carve(BT * DI * 4);
    _Float16* xc16    = (_Float16*)carve(BT * DI * 2);
    float*    xdbl    = (float*)   carve(BT * XPE * 4);
    _Float16* xdbl16  = (_Float16*)carve(BT * XPE * 2);
    float*    dtbuf   = (float*)   carve(BT * DI * 4);
    _Float16* y16     = (_Float16*)carve(BT * DI * 2);

    auto cvt = [&](const float* src, _Float16* dst, int n) {
        cvt_f32_to_f16<<<(n + 255) / 256, 256, 0, stream>>>(src, dst, n);
    };

    // one-time f16 weight / input conversions
    cvt(x, x16, (int)(BT * FF));
    cvt(proj_w, wproj16, DM * FF);
    cvt(ipw, win16, LL * 2 * DI * DM);
    cvt(xpw, wxp16, LL * XPE * DI);
    cvt(dtw, wdt16, LL * DI * DTR);
    cvt(opw, wout16, LL * DM * DI);

    const int Mstripes = (int)(BT / 32);  // 256 (32 rows per wave)
    // grid.y covers n-tile PAIRS, 4 waves (pairs) per block
    auto gy = [](int ntiles) { return (((ntiles + 1) / 2) + 3) / 4; };

    // input projection: h = x @ proj_w^T + proj_b   (M=8192,K=64,N=256)
    gemm_f16_wmma<false, 1, true><<<dim3(Mstripes, gy(DM / 16)), 128, 0, stream>>>(
        x16, FF, wproj16, FF, hbuf, h16, DM,
        DM / 16, FF, FF, proj_b);

    for (int l = 0; l < LL; ++l) {
        const _Float16* ipw_l = win16  + (size_t)l * 2 * DI * DM;
        const _Float16* xpw_l = wxp16  + (size_t)l * XPE * DI;
        const _Float16* dtw_l = wdt16  + (size_t)l * DI * DTR;
        const _Float16* opw_l = wout16 + (size_t)l * DM * DI;
        const float* cw_l  = conv_w + (size_t)l * DI * DCV;
        const float* cb_l  = conv_b + (size_t)l * DI;
        const float* dtb_l = dtb    + (size_t)l * DI;
        const float* Al_l  = Alog   + (size_t)l * DI * NN;
        const float* Dp_l  = Dp     + (size_t)l * DI;

        // in_proj: xz = h @ ipw^T   (K=256, N=1024)
        gemm_f16_wmma<false, 0, false><<<dim3(Mstripes, gy(2 * DI / 16)), 128, 0, stream>>>(
            h16, DM, ipw_l, DM, xzbuf, nullptr, 2 * DI,
            2 * DI / 16, DM, DM, nullptr);

        // causal depthwise conv + silu
        conv_silu_kernel<<<(int)(BT * DI / 256), 256, 0, stream>>>(
            xzbuf, cw_l, cb_l, xcbuf, xc16);

        // x_proj: xdbl = xc @ xpw^T  (K=512, N=48 -> 3 tiles)
        gemm_f16_wmma<false, 0, true><<<dim3(Mstripes, gy(XPE / 16)), 128, 0, stream>>>(
            xc16, DI, xpw_l, DI, xdbl, xdbl16, XPE,
            XPE / 16, DI, DI, nullptr);

        // dt_proj: dt = softplus(xdbl[:, :16] @ dtw^T + dtb)  (K=16 padded to 32)
        gemm_f16_wmma<true, 2, false><<<dim3(Mstripes, gy(DI / 16)), 128, 0, stream>>>(
            xdbl16, XPE, dtw_l, DTR, dtbuf, nullptr, DI,
            DI / 16, 32, DTR, dtb_l);

        // selective scan (+ D*u skip, silu(z) gate) -> y16
        scan_kernel<<<BB, DI, 0, stream>>>(
            dtbuf, xdbl, xcbuf, xzbuf, Al_l, Dp_l, y16);

        // out_proj: h = y @ opw^T  (K=512, N=256)
        gemm_f16_wmma<false, 0, true><<<dim3(Mstripes, gy(DM / 16)), 128, 0, stream>>>(
            y16, DI, opw_l, DI, hbuf, h16, DM,
            DM / 16, DI, DI, nullptr);
    }

    // layernorm on last token + linear head
    ln_head_kernel<<<BB, DM, 0, stream>>>(hbuf, ln_g, ln_b, head_w, head_b, out);
}
